// PhysicsInformedAttention_4372276707448
// MI455X (gfx1250) — matrix operand order repaired and checked by reading, not compiled, not run on hardware
//
#include <hip/hip_runtime.h>
#include <hip/hip_bf16.h>

#define B_   2
#define N_   2048
#define C_   1024
#define H_   16
#define DH_  64

typedef __attribute__((ext_vector_type(16))) __bf16 bf16x16;
typedef __attribute__((ext_vector_type(8)))  float  f32x8;

union BF16FragU { bf16x16 v; uint4 u[2]; };
union PackU     { uint4 u; __bf16 e[8]; };

__device__ __forceinline__ f32x8 zero8() {
    f32x8 z = {0.f,0.f,0.f,0.f,0.f,0.f,0.f,0.f};
    return z;
}

// A-matrix fragment (16x32 bf16) from an f32 row-major source.
// Lane half h holds K = h*8 + [0..7] and 16 + h*8 + [0..7]; p at (row, k0+half*8).
__device__ __forceinline__ bf16x16 load_a_f32(const float* __restrict__ p) {
    float4 c0 = *(const float4*)(p);
    float4 c1 = *(const float4*)(p + 4);
    float4 c2 = *(const float4*)(p + 16);
    float4 c3 = *(const float4*)(p + 20);
    bf16x16 r;
    r[0]=(__bf16)c0.x;  r[1]=(__bf16)c0.y;  r[2]=(__bf16)c0.z;  r[3]=(__bf16)c0.w;
    r[4]=(__bf16)c1.x;  r[5]=(__bf16)c1.y;  r[6]=(__bf16)c1.z;  r[7]=(__bf16)c1.w;
    r[8]=(__bf16)c2.x;  r[9]=(__bf16)c2.y;  r[10]=(__bf16)c2.z; r[11]=(__bf16)c2.w;
    r[12]=(__bf16)c3.x; r[13]=(__bf16)c3.y; r[14]=(__bf16)c3.z; r[15]=(__bf16)c3.w;
    return r;
}

// B-matrix fragment (32x16 bf16) from f32: K = h*8 + [0..7] and 16 + h*8 + [0..7]; p at (col, k0+half*8).
__device__ __forceinline__ bf16x16 load_b_f32(const float* __restrict__ p) {
    float4 c0 = *(const float4*)(p);
    float4 c1 = *(const float4*)(p + 4);
    float4 c2 = *(const float4*)(p + 16);
    float4 c3 = *(const float4*)(p + 20);
    bf16x16 r;
    r[0]=(__bf16)c0.x;  r[1]=(__bf16)c0.y;  r[2]=(__bf16)c0.z;  r[3]=(__bf16)c0.w;
    r[4]=(__bf16)c1.x;  r[5]=(__bf16)c1.y;  r[6]=(__bf16)c1.z;  r[7]=(__bf16)c1.w;
    r[8]=(__bf16)c2.x;  r[9]=(__bf16)c2.y;  r[10]=(__bf16)c2.z; r[11]=(__bf16)c2.w;
    r[12]=(__bf16)c3.x; r[13]=(__bf16)c3.y; r[14]=(__bf16)c3.z; r[15]=(__bf16)c3.w;
    return r;
}

// C = A(MxK,f32) @ W(NxK,f32)^T + bias.
// MODE 0: write f32 row-major to outF.
// MODE 1: QKV epilogue -> scatter bf16 into q, k (B,H,N,Dh) and vT (B,H,Dh,N).
template<int MODE, int M, int N, int K>
__global__ void __launch_bounds__(256)
gemm_bf16_wmma(const float* __restrict__ A,
               const float* __restrict__ W,
               const float* __restrict__ bias,
               float* __restrict__ outF,
               __bf16* __restrict__ qOut,
               __bf16* __restrict__ kOut,
               __bf16* __restrict__ vTOut)
{
    const int lane = threadIdx.x & 31;
    const int wid  = blockIdx.x * (blockDim.x >> 5) + (threadIdx.x >> 5);
    const int tilesN = N / 64;
    const int tm = wid / tilesN;
    const int tn = wid % tilesN;
    if (tm >= M / 64) return;
    const int lc   = lane & 15;
    const int half = lane >> 4;
    const int rowBase = tm * 64;
    const int colBase = tn * 64;

    f32x8 acc[4][4];
#pragma unroll
    for (int i = 0; i < 4; ++i)
#pragma unroll
        for (int j = 0; j < 4; ++j) acc[i][j] = zero8();

#pragma unroll 1
    for (int k0 = 0; k0 < K; k0 += 32) {
        bf16x16 a[4], b[4];
#pragma unroll
        for (int i = 0; i < 4; ++i)
            a[i] = load_a_f32(A + (size_t)(rowBase + 16*i + lc) * K + k0 + half*8);
#pragma unroll
        for (int j = 0; j < 4; ++j)
            b[j] = load_b_f32(W + (size_t)(colBase + 16*j + lc) * K + k0 + half*8);
#pragma unroll
        for (int i = 0; i < 4; ++i)
#pragma unroll
            for (int j = 0; j < 4; ++j)
                acc[i][j] = __builtin_amdgcn_wmma_f32_16x16x32_bf16(
                    false, a[i], false, b[j], (short)0, acc[i][j], false, false);
    }

#pragma unroll
    for (int i = 0; i < 4; ++i) {
        const int row0 = rowBase + 16*i + half*8;
#pragma unroll
        for (int j = 0; j < 4; ++j) {
            const int col = colBase + 16*j + lc;
            const float bv = bias[col];
#pragma unroll
            for (int r = 0; r < 8; ++r) {
                const float val = acc[i][j][r] + bv;
                const int row = row0 + r;
                if (MODE == 0) {
                    outF[(size_t)row * N + col] = val;
                } else {
                    const int s  = col >> 10;          // 0=q 1=k 2=v
                    const int h  = (col >> 6) & (H_ - 1);
                    const int d  = col & (DH_ - 1);
                    const int bb = row >> 11;          // / N_
                    const int nn = row & (N_ - 1);
                    const __bf16 bw = (__bf16)val;
                    if (s == 0)
                        qOut[(((size_t)bb * H_ + h) * N_ + nn) * DH_ + d] = bw;
                    else if (s == 1)
                        kOut[(((size_t)bb * H_ + h) * N_ + nn) * DH_ + d] = bw;
                    else
                        vTOut[(((size_t)bb * H_ + h) * DH_ + d) * N_ + nn] = bw;
                }
            }
        }
    }
}

// Flash attention, transposed formulation. One wave per 32 query rows.
// S^T = K @ Q^T  => queries in lanes, keys in VGPRs:
//   - softmax reduction over keys = in-lane over 16 VGPR values + ONE shfl_xor(16)
//   - m/l/alpha are per-lane scalars
//   - mask loads are int4 vectors (8 contiguous keys per lane)
// O^T = V^T @ P^T accumulated per q-tile; V A-fragments shared by both q-tiles.
// launch_bounds(256,1): full VGPR budget -> no spills.
__global__ void __launch_bounds__(256, 1)
flash_attn_wmma(const __bf16* __restrict__ Q,
                const __bf16* __restrict__ Kb,
                const __bf16* __restrict__ VT,
                const int*    __restrict__ mask,
                const float*  __restrict__ pbias,
                float* __restrict__ Out)
{
    __shared__ __bf16 ldsP[8 * 2 * 16 * 32];   // per-wave: two P^T tiles [q][key]

    const int lane  = threadIdx.x & 31;
    const int wslot = threadIdx.x >> 5;
    const int wid   = blockIdx.x * 8 + wslot;
    const int qb = wid & (N_/32 - 1);          // 64 q-tiles of 32 rows
    const int bh = wid / (N_/32);
    const int h  = bh & (H_ - 1);
    const int b  = bh / H_;
    const int lc   = lane & 15;
    const int half = lane >> 4;
    const float scale = 0.125f;                // Dh^-0.5
    const float pb = pbias[h];

    const __bf16* qBase = Q  + (size_t)bh * N_ * DH_;
    const int qrow0 = qb * 32;

    // Q fragments as B-matrix (Q^T): lane = query, dh = half*8 + [0..7] and 16 + half*8 + [0..7].
    BF16FragU qf[2][2];
#pragma unroll
    for (int qt = 0; qt < 2; ++qt)
#pragma unroll
        for (int kk = 0; kk < 2; ++kk) {
            const __bf16* p = qBase + (size_t)(qrow0 + qt*16 + lc) * DH_ + kk*32 + half*8;
            qf[qt][kk].u[0] = *(const uint4*)(p);
            qf[qt][kk].u[1] = *(const uint4*)(p + 16);
        }

    // Loop-carried per-lane pointers (all inner offsets are constants)
    const __bf16* kPtr = Kb + (size_t)bh * N_ * DH_ + (size_t)lc * DH_ + half*8;
    const __bf16* vPtr = VT + (size_t)bh * DH_ * N_ + (size_t)lc * N_ + half*8;
    const int*    mPtr0 = mask + (size_t)b * N_ * N_
                               + (size_t)(qrow0 + lc) * N_ + half*8;
    const int*    mPtr1 = mPtr0 + 16 * N_;

    f32x8 o[2][4];                 // O^T tiles: vgpr = dh row, lane = query
    float m[2], l[2];
#pragma unroll
    for (int qt = 0; qt < 2; ++qt) {
#pragma unroll
        for (int d = 0; d < 4; ++d) o[qt][d] = zero8();
        m[qt] = -1e30f; l[qt] = 0.f;
    }

    __bf16* myP = ldsP + wslot * (2 * 16 * 32);

#pragma unroll 1
    for (int kb = 0; kb < N_ / 32; ++kb) {
        __builtin_prefetch(kPtr + 32*DH_, 0, 1);   // next key block

        // ---- Phase 1: S^T tiles + online softmax, P^T -> LDS ----
        {
            BF16FragU kf[2][2];    // A-matrix: rows = keys (live only here)
#pragma unroll
            for (int jj = 0; jj < 2; ++jj)
#pragma unroll
                for (int kk = 0; kk < 2; ++kk) {
                    const __bf16* p = kPtr + jj*16*DH_ + kk*32;
                    kf[jj][kk].u[0] = *(const uint4*)(p);
                    kf[jj][kk].u[1] = *(const uint4*)(p + 16);
                }

#pragma unroll
            for (int qt = 0; qt < 2; ++qt) {
                // S^T tiles: vgpr r = key (r + 8*half + 16*jj), lane = query
                float sv[2][8];
#pragma unroll
                for (int jj = 0; jj < 2; ++jj) {
                    f32x8 accs = zero8();
#pragma unroll
                    for (int kk = 0; kk < 2; ++kk)
                        accs = __builtin_amdgcn_wmma_f32_16x16x32_bf16(
                            false, kf[jj][kk].v, false, qf[qt][kk].v, (short)0, accs, false, false);
#pragma unroll
                    for (int r = 0; r < 8; ++r)
                        sv[jj][r] = accs[r] * scale + pb;
                }

                // in-lane max over 16 keys + one half-merge shuffle
                float mx = sv[0][0];
#pragma unroll
                for (int r = 1; r < 8; ++r) mx = fmaxf(mx, sv[0][r]);
#pragma unroll
                for (int r = 0; r < 8; ++r) mx = fmaxf(mx, sv[1][r]);
                mx = fmaxf(mx, __shfl_xor(mx, 16, 32));
                const float mn    = fmaxf(m[qt], mx);
                const float alpha = __expf(m[qt] - mn);
                m[qt] = mn;

                // P = exp(s-mn)*mask, packed 8xbf16 -> one ds_store_b128 per tile
                const int* mp = qt ? mPtr1 : mPtr0;
                __bf16* pTile = myP + qt * (16*32);
                float rs = 0.f;
#pragma unroll
                for (int jj = 0; jj < 2; ++jj) {
                    // mask: 2 x int4, contiguous keys for this lane's query
                    const int4 mlo = *(const int4*)(mp + jj*16);
                    const int4 mhi = *(const int4*)(mp + jj*16 + 4);
                    PackU pk;
                    float p0 = __expf(sv[jj][0] - mn) * (float)mlo.x;
                    float p1 = __expf(sv[jj][1] - mn) * (float)mlo.y;
                    float p2 = __expf(sv[jj][2] - mn) * (float)mlo.z;
                    float p3 = __expf(sv[jj][3] - mn) * (float)mlo.w;
                    float p4 = __expf(sv[jj][4] - mn) * (float)mhi.x;
                    float p5 = __expf(sv[jj][5] - mn) * (float)mhi.y;
                    float p6 = __expf(sv[jj][6] - mn) * (float)mhi.z;
                    float p7 = __expf(sv[jj][7] - mn) * (float)mhi.w;
                    rs += ((p0+p1)+(p2+p3)) + ((p4+p5)+(p6+p7));
                    pk.e[0]=(__bf16)p0; pk.e[1]=(__bf16)p1;
                    pk.e[2]=(__bf16)p2; pk.e[3]=(__bf16)p3;
                    pk.e[4]=(__bf16)p4; pk.e[5]=(__bf16)p5;
                    pk.e[6]=(__bf16)p6; pk.e[7]=(__bf16)p7;
                    *(uint4*)(pTile + lc*32 + jj*16 + half*8) = pk.u;
                }
                rs += __shfl_xor(rs, 16, 32);
                l[qt] = l[qt] * alpha + rs;
#pragma unroll
                for (int d = 0; d < 4; ++d)
#pragma unroll
                    for (int r = 0; r < 8; ++r) o[qt][d][r] *= alpha;
            }
        }

        // ---- Phase 2: O^T += V^T @ P^T. Each V fragment feeds both q-tiles. ----
        BF16FragU pf[2];           // B-matrix: lane = query, keys half*8 + [0..7] and 16 + half*8 + [0..7]
#pragma unroll
        for (int qt = 0; qt < 2; ++qt) {
            const __bf16* p = myP + qt * (16*32) + lc*32 + half*8;
            pf[qt].u[0] = *(const uint4*)(p);
            pf[qt].u[1] = *(const uint4*)(p + 16);
        }
#pragma unroll
        for (int d = 0; d < 4; ++d) {
            BF16FragU vf;          // A-matrix: rows = dh, K = keys
            const __bf16* p = vPtr + d*16*N_;
            vf.u[0] = *(const uint4*)(p);
            vf.u[1] = *(const uint4*)(p + 16);
            o[0][d] = __builtin_amdgcn_wmma_f32_16x16x32_bf16(
                false, vf.v, false, pf[0].v, (short)0, o[0][d], false, false);
            o[1][d] = __builtin_amdgcn_wmma_f32_16x16x32_bf16(
                false, vf.v, false, pf[1].v, (short)0, o[1][d], false, false);
        }

        kPtr  += 32 * DH_;   // next 32 key rows
        vPtr  += 32;         // next 32 key columns of V^T
        mPtr0 += 32;
        mPtr1 += 32;
    }

    // normalize and write ctx: O^T element (dh = d*16 + r + 8*half, q = lc);
    // 8 contiguous f32 per (lane,d) -> mergeable stores
#pragma unroll
    for (int qt = 0; qt < 2; ++qt) {
        const float inv = 1.0f / l[qt];
        const int n = qrow0 + qt*16 + lc;
        float* dst = Out + ((size_t)b * N_ + n) * C_ + h * DH_ + half*8;
#pragma unroll
        for (int d = 0; d < 4; ++d)
#pragma unroll
            for (int r = 0; r < 8; ++r)
                dst[d*16 + r] = o[qt][d][r] * inv;
    }
}

extern "C" void kernel_launch(void* const* d_in, const int* in_sizes, int n_in,
                              void* d_out, int out_size, void* d_ws, size_t ws_size,
                              hipStream_t stream)
{
    const float* x      = (const float*)d_in[0];
    const float* qkv_w  = (const float*)d_in[1];
    const float* qkv_b  = (const float*)d_in[2];
    const float* proj_w = (const float*)d_in[3];
    const float* proj_b = (const float*)d_in[4];
    const float* pbias  = (const float*)d_in[5];
    const int*   mask   = (const int*)d_in[6];
    float* out = (float*)d_out;

    const size_t headElems = (size_t)B_ * H_ * N_ * DH_;   // 4Mi
    __bf16* qb  = (__bf16*)d_ws;
    __bf16* kb  = qb + headElems;
    __bf16* vT  = kb + headElems;
    float*  ctx = (float*)(vT + headElems);                // (B,N,C) f32

    // 1) fused QKV projection: (4096x1024) @ (3072x1024)^T + b
    {
        const int tiles = (B_*N_/64) * (3*C_/64);          // 3072 wave-tiles
        gemm_bf16_wmma<1, B_*N_, 3*C_, C_><<<tiles/8, 256, 0, stream>>>(
            x, qkv_w, qkv_b, nullptr, qb, kb, vT);
    }
    // 2) flash attention with physics bias + mask
    {
        const int waves = B_ * H_ * (N_/32);               // 2048 waves
        flash_attn_wmma<<<waves/8, 256, 0, stream>>>(qb, kb, vT, mask, pbias, ctx);
    }
    // 3) output projection: (4096x1024) @ (1024x1024)^T + b -> d_out
    {
        const int tiles = (B_*N_/64) * (C_/64);            // 1024 wave-tiles
        gemm_bf16_wmma<0, B_*N_, C_, C_><<<tiles/8, 256, 0, stream>>>(
            ctx, proj_w, proj_b, out, nullptr, nullptr, nullptr);
    }
}
